// ExperimentalGNN_36060545417566
// MI455X (gfx1250) — matrix operand order repaired and checked by reading, not compiled, not run on hardware
//
#include <hip/hip_runtime.h>
#include <hip/hip_bf16.h>
#include <math.h>
#include <utility>

typedef __attribute__((ext_vector_type(16))) _Float16 v16h;
typedef __attribute__((ext_vector_type(8)))  float    v8f;
typedef __attribute__((ext_vector_type(4)))  unsigned u32x4;
typedef __attribute__((ext_vector_type(8)))  int      i32x8;
typedef __attribute__((ext_vector_type(4)))  int      i32x4;

#define HD      256
#define NNODES  8192
#define NEDGES  131072
#define NGRAPHS 64
#define NHEADS  8
#define DHEAD   32

#define EPI_NONE    0
#define EPI_BN_SILU 1

__device__ __forceinline__ float sigm_(float x) { return 1.0f / (1.0f + __expf(-x)); }
__device__ __forceinline__ float silu_(float x) { return x / (1.0f + __expf(-x)); }

__device__ __forceinline__ void atomicMaxF(float* addr, float val) {
    if (val >= 0.0f) atomicMax((int*)addr, __float_as_int(val));
    else             atomicMin((unsigned int*)addr, (unsigned int)__float_as_int(val));
}

// ---------------------------------------------------------------------------
// TDM: 1-row tensor (nelem f32) global -> LDS, issued by the calling wave.
// D# built per CDNA5 ISA:
//   group0: count=1 | lds_addr[63:32] | global_addr[120:64] | type=2
//   group1: wg_mask=0, data_size=4B, tensor_dim0=nelem, tensor_dim1=1,
//           tile_dim0=nelem, tile_dim1=1, dim0_stride=nelem
// Completion via TENSORcnt (s_wait_tensorcnt 0).
// ---------------------------------------------------------------------------
#if __has_builtin(__builtin_amdgcn_tensor_load_to_lds)
#define HAVE_TDM 1
__device__ __forceinline__ void tdm_load_1d_f32(unsigned lds_byte, const float* gsrc,
                                                unsigned nelem) {
    unsigned long long ga = (unsigned long long)gsrc;
    u32x4 g0;
    g0[0] = 1u;                                                   // count=1
    g0[1] = lds_byte;                                             // lds_addr
    g0[2] = (unsigned)(ga & 0xffffffffull);                       // gaddr[95:64]
    g0[3] = (unsigned)((ga >> 32) & 0x1ffffffull) | (2u << 30);   // gaddr hi | type=2
    i32x8 g1;
    g1[0] = (int)(2u << 16);                      // workgroup_mask=0, data_size=4B
    g1[1] = (int)((nelem & 0xffffu) << 16);       // tensor_dim0[15:0]
    g1[2] = (int)(((nelem >> 16) & 0xffffu) | (1u << 16)); // dim0 hi | tensor_dim1=1
    g1[3] = (int)((nelem & 0xffffu) << 16);       // tensor_dim1 hi=0 | tile_dim0
    g1[4] = 1;                                    // tile_dim1=1, tile_dim2=0
    g1[5] = (int)nelem;                           // tensor_dim0_stride lo
    g1[6] = 0;
    g1[7] = 0;
    i32x4 gz = {0, 0, 0, 0};
#if defined(__clang_major__) && (__clang_major__ >= 23)
    i32x8 gz8 = {0, 0, 0, 0, 0, 0, 0, 0};
    __builtin_amdgcn_tensor_load_to_lds(g0, g1, gz, gz, gz8, 0);
#else
    __builtin_amdgcn_tensor_load_to_lds(g0, g1, gz, gz, 0);
#endif
}
#else
#define HAVE_TDM 0
#endif

// ---------------------------------------------------------------------------
// WMMA GEMM:  C[M,N] = epi( (A[M,K] (*rowscale)) @ W[K,N] + bias )
// Tiles: BM=128, BN=128, BK=32. 256 threads = 8 wave32 waves.
// Wave w owns rows 16w..16w+15 across 128 cols: 8 x 16x16 WMMA tiles, so each
// A fragment is reused by 8 v_wmma_f32_16x16x32_f16 ops. fp32 -> f16 during
// LDS staging; double-buffered LDS, one barrier per K-step.
// ---------------------------------------------------------------------------
#define GBM 128
#define GBN 128
#define GBK 32
#define LKP (GBK + 8)   // LDS row pitch in halves (80 bytes, 16B aligned)

template <bool HAS_RS, int MODE>
__global__ __launch_bounds__(256)
void wmma_gemm_t(const float* __restrict__ A, const float* __restrict__ W,
                 const float* __restrict__ bias,
                 const float* __restrict__ bn_s, const float* __restrict__ bn_o,
                 const float* __restrict__ rowscale,
                 float* __restrict__ C, int M, int N, int K)
{
    __shared__ alignas(16) _Float16 As[2][GBM][LKP];
    __shared__ alignas(16) _Float16 Wt[2][GBN][LKP];  // transposed (n, k)

    const int tid  = threadIdx.x;
    const int wave = tid >> 5;
    const int lane = tid & 31;
    const int bm   = blockIdx.x * GBM;
    const int bn   = blockIdx.y * GBN;

    // A staging: thread t handles row t/2, 16-col half (t&1); 4x float4 load,
    // f16 convert, 2x ds_store_b128.
    const int arow = tid >> 1;
    const int acol = (tid & 1) << 4;
    auto stageA = [&](int ks, int buf) {
        const float* ap = A + (size_t)(bm + arow) * K + (ks + acol);
        float4 a0 = *(const float4*)(ap + 0);
        float4 a1 = *(const float4*)(ap + 4);
        float4 a2 = *(const float4*)(ap + 8);
        float4 a3 = *(const float4*)(ap + 12);
        float sc = 1.0f;
        if (HAS_RS) sc = rowscale[bm + arow];
        union { _Float16 h[16]; uint4 u4[2]; } pk;
        pk.h[0]  = (_Float16)(a0.x * sc); pk.h[1]  = (_Float16)(a0.y * sc);
        pk.h[2]  = (_Float16)(a0.z * sc); pk.h[3]  = (_Float16)(a0.w * sc);
        pk.h[4]  = (_Float16)(a1.x * sc); pk.h[5]  = (_Float16)(a1.y * sc);
        pk.h[6]  = (_Float16)(a1.z * sc); pk.h[7]  = (_Float16)(a1.w * sc);
        pk.h[8]  = (_Float16)(a2.x * sc); pk.h[9]  = (_Float16)(a2.y * sc);
        pk.h[10] = (_Float16)(a2.z * sc); pk.h[11] = (_Float16)(a2.w * sc);
        pk.h[12] = (_Float16)(a3.x * sc); pk.h[13] = (_Float16)(a3.y * sc);
        pk.h[14] = (_Float16)(a3.z * sc); pk.h[15] = (_Float16)(a3.w * sc);
        *(uint4*)&As[buf][arow][acol]     = pk.u4[0];
        *(uint4*)&As[buf][arow][acol + 8] = pk.u4[1];
    };
    // W staging (transposed): item idx -> (col nn, k-pair kp); two coalesced
    // b32 loads, packed f16x2 b32 store. 32x128 elements per tile.
    auto stageW = [&](int ks, int buf) {
#pragma unroll
        for (int it = 0; it < 8; ++it) {
            int idx = tid + it * 256;
            int nn  = idx & 127;
            int kp  = (idx >> 7) << 1;
            float w0 = W[(size_t)(ks + kp) * N + (bn + nn)];
            float w1 = W[(size_t)(ks + kp + 1) * N + (bn + nn)];
            union { _Float16 h[2]; unsigned u; } p;
            p.h[0] = (_Float16)w0;
            p.h[1] = (_Float16)w1;
            *(unsigned*)&Wt[buf][nn][kp] = p.u;
        }
    };

    const v8f vzero = {};
    v8f acc[8] = {vzero, vzero, vzero, vzero, vzero, vzero, vzero, vzero};

    stageA(0, 0);
    stageW(0, 0);
    __syncthreads();

    int cur = 0;
    for (int ks = 0; ks < K; ks += GBK) {
        // issue next tile's loads first (overlaps with WMMA below)
        if (ks + GBK < K) {
            stageA(ks + GBK, cur ^ 1);
            stageW(ks + GBK, cur ^ 1);
            if (ks + 2 * GBK < K)
                __builtin_prefetch(&A[(size_t)(bm + arow) * K + (ks + 2 * GBK)], 0, 1);
        }

        // A fragment, ISA 16-bit A 16x32 layout:
        // lane L: row m=L%16; VGPR v pairs at k0=(v>=4)*16 + (L/16)*8 + (v%4)*2
        v16h af;
        {
            int m    = (wave << 4) + (lane & 15);
            int half = lane >> 4;
#pragma unroll
            for (int v = 0; v < 8; ++v) {
                int k0 = ((v >> 2) << 4) + (half << 3) + ((v & 3) << 1);
                af[2 * v]     = As[cur][m][k0];
                af[2 * v + 1] = As[cur][m][k0 + 1];
            }
        }
        // B fragment, ISA B 32x16 layout: lane L: col n=L%16; k=(v>=4)*16+(L/16)*8+(v%4)*2
#pragma unroll
        for (int nt = 0; nt < 8; ++nt) {
            v16h bf;
            int n  = (nt << 4) + (lane & 15);
            int kb = (lane >> 4) << 3;
#pragma unroll
            for (int v = 0; v < 8; ++v) {
                int kk = ((v >> 2) << 4) + kb + ((v & 3) << 1);
                bf[2 * v]     = Wt[cur][n][kk];
                bf[2 * v + 1] = Wt[cur][n][kk + 1];
            }
            acc[nt] = __builtin_amdgcn_wmma_f32_16x16x32_f16(
                false, af, false, bf, (short)0, acc[nt], false, false);
        }
        __syncthreads();
        cur ^= 1;
    }

    // Epilogue. C/D layout: VGPR r -> row (lane/16)*8 + r, col = lane%16.
    const int mbase = bm + (wave << 4) + ((lane >> 4) << 3);
#pragma unroll
    for (int nt = 0; nt < 8; ++nt) {
        int col = bn + (nt << 4) + (lane & 15);
        float b = bias ? bias[col] : 0.0f;
        float s = 1.0f, o = 0.0f;
        if (MODE == EPI_BN_SILU) { s = bn_s[col]; o = bn_o[col]; }
#pragma unroll
        for (int r = 0; r < 8; ++r) {
            float y = acc[nt][r] + b;
            if (MODE == EPI_BN_SILU) { y = y * s + o; y = silu_(y); }
            C[(size_t)(mbase + r) * N + col] = y;
        }
    }
}

// ---------------------------------------------------------------------------
// Small / irregular kernels
// ---------------------------------------------------------------------------
__global__ void fill_kernel(float* __restrict__ p, float v, long long n) {
    long long i = (long long)blockIdx.x * blockDim.x + threadIdx.x;
    if (i < n) p[i] = v;
}

// tiny-K encoder: C = silu(bn(A@W + b)), thread per (m,n)
__global__ void enc_kernel(const float* __restrict__ A, const float* __restrict__ W,
                           const float* __restrict__ b, const float* __restrict__ s,
                           const float* __restrict__ o, float* __restrict__ C,
                           int M, int K, int N) {
    long long idx = (long long)blockIdx.x * blockDim.x + threadIdx.x;
    if (idx >= (long long)M * N) return;
    int m = (int)(idx / N), n = (int)(idx % N);
    float acc = b[n];
    for (int k = 0; k < K; ++k) acc += A[(size_t)m * K + k] * W[(size_t)k * N + n];
    acc = acc * s[n] + o[n];
    C[idx] = silu_(acc);
}

// w[m] = sigmoid(dot(T[m,:], W2) + b2), K = HD
__global__ void att_w_kernel(const float* __restrict__ T, const float* __restrict__ W2,
                             const float* __restrict__ b2, float* __restrict__ w, int M) {
    int m = blockIdx.x * blockDim.x + threadIdx.x;
    if (m >= M) return;
    const float4* t4 = (const float4*)(T + (size_t)m * HD);
    const float4* w4 = (const float4*)W2;
    float acc = b2[0];
    for (int k = 0; k < HD / 4; ++k) {
        float4 a = t4[k], bb = w4[k];
        acc += a.x * bb.x + a.y * bb.y + a.z * bb.z + a.w * bb.w;
    }
    w[m] = sigm_(acc);
}

// GINE: agg[dst] += relu(h[src] + e*w), 4 cols per thread
__global__ void gine_scatter_kernel(const float* __restrict__ h, const float* __restrict__ e,
                                    const float* __restrict__ wE, const int* __restrict__ src,
                                    const int* __restrict__ dst, float* __restrict__ agg) {
    long long idx = (long long)blockIdx.x * blockDim.x + threadIdx.x;
    if (idx >= (long long)NEDGES * (HD / 4)) return;
    int edge = (int)(idx >> 6);
    int c4   = (int)(idx & 63) << 2;
    int sj = src[edge], di = dst[edge];
    float we = wE[edge];
    float4 hv = *(const float4*)(h + (size_t)sj * HD + c4);
    float4 ev = *(const float4*)(e + (size_t)edge * HD + c4);
    float* ap = agg + (size_t)di * HD + c4;
    atomicAdd(ap + 0, fmaxf(hv.x + ev.x * we, 0.0f));
    atomicAdd(ap + 1, fmaxf(hv.y + ev.y * we, 0.0f));
    atomicAdd(ap + 2, fmaxf(hv.z + ev.z * we, 0.0f));
    atomicAdd(ap + 3, fmaxf(hv.w + ev.w * we, 0.0f));
}

__global__ void gine_combine_kernel(const float* __restrict__ h, const float* __restrict__ agg,
                                    const float* __restrict__ eps, float* __restrict__ z,
                                    long long n) {
    long long i = (long long)blockIdx.x * blockDim.x + threadIdx.x;
    if (i < n) z[i] = (1.0f + eps[0]) * h[i] + agg[i];
}

// h += hn * s[col] + o[col]
__global__ void resid_bn_kernel(float* __restrict__ h, const float* __restrict__ hn,
                                const float* __restrict__ s, const float* __restrict__ o,
                                long long n) {
    long long i = (long long)blockIdx.x * blockDim.x + threadIdx.x;
    if (i >= n) return;
    int col = (int)(i & (HD - 1));
    h[i] += hn[i] * s[col] + o[col];
}

// alpha[e,head] = dot(q[dst,head], k[src,head] + eh[e,head]) / sqrt(DH)
__global__ void tf_alpha_kernel(const float* __restrict__ q, const float* __restrict__ k,
                                const float* __restrict__ eh, const int* __restrict__ src,
                                const int* __restrict__ dst, float* __restrict__ alpha) {
    long long idx = (long long)blockIdx.x * blockDim.x + threadIdx.x;
    if (idx >= (long long)NEDGES * NHEADS) return;
    int edge = (int)(idx >> 3), head = (int)(idx & 7);
    int sj = src[edge], di = dst[edge];
    const float4* qp = (const float4*)(q + (size_t)di * HD + head * DHEAD);
    const float4* kp = (const float4*)(k + (size_t)sj * HD + head * DHEAD);
    const float4* ep = (const float4*)(eh + (size_t)edge * HD + head * DHEAD);
    float acc = 0.0f;
    for (int j = 0; j < DHEAD / 4; ++j) {
        float4 qv = qp[j], kv = kp[j], ev = ep[j];
        acc += qv.x * (kv.x + ev.x) + qv.y * (kv.y + ev.y) +
               qv.z * (kv.z + ev.z) + qv.w * (kv.w + ev.w);
    }
    alpha[idx] = acc * 0.17677669529663687f;  // 1/sqrt(32)
}

__global__ void seg_max_kernel(const float* __restrict__ alpha, const int* __restrict__ dst,
                               float* __restrict__ m) {
    long long idx = (long long)blockIdx.x * blockDim.x + threadIdx.x;
    if (idx >= (long long)NEDGES * NHEADS) return;
    int edge = (int)(idx >> 3), head = (int)(idx & 7);
    atomicMaxF(&m[(size_t)dst[edge] * NHEADS + head], alpha[idx]);
}

// in-place: alpha <- exp(alpha - m[dst]); den[dst] += alpha
__global__ void seg_exp_kernel(float* __restrict__ alpha, const int* __restrict__ dst,
                               const float* __restrict__ m, float* __restrict__ den) {
    long long idx = (long long)blockIdx.x * blockDim.x + threadIdx.x;
    if (idx >= (long long)NEDGES * NHEADS) return;
    int edge = (int)(idx >> 3), head = (int)(idx & 7);
    float ex = __expf(alpha[idx] - m[(size_t)dst[edge] * NHEADS + head]);
    alpha[idx] = ex;
    atomicAdd(&den[(size_t)dst[edge] * NHEADS + head], ex);
}

// out[dst] += (v[src] + eh) * att
__global__ void tf_out_kernel(const float* __restrict__ v, const float* __restrict__ eh,
                              const float* __restrict__ ex, const float* __restrict__ den,
                              const int* __restrict__ src, const int* __restrict__ dst,
                              float* __restrict__ out) {
    long long idx = (long long)blockIdx.x * blockDim.x + threadIdx.x;
    if (idx >= (long long)NEDGES * (HD / 4)) return;
    int edge = (int)(idx >> 6);
    int c4   = (int)(idx & 63) << 2;
    int head = c4 >> 5;
    int sj = src[edge], di = dst[edge];
    float att = ex[(size_t)edge * NHEADS + head] /
                (den[(size_t)di * NHEADS + head] + 1e-16f);
    float4 vv = *(const float4*)(v + (size_t)sj * HD + c4);
    float4 ee = *(const float4*)(eh + (size_t)edge * HD + c4);
    float* op = out + (size_t)di * HD + c4;
    atomicAdd(op + 0, (vv.x + ee.x) * att);
    atomicAdd(op + 1, (vv.y + ee.y) * att);
    atomicAdd(op + 2, (vv.z + ee.z) * att);
    atomicAdd(op + 3, (vv.w + ee.w) * att);
}

// beta = sigmoid([out, xr, out-xr] . Wb); Wb staged to LDS via TDM
__global__ __launch_bounds__(256)
void tf_beta_kernel(const float* __restrict__ out, const float* __restrict__ xr,
                    const float* __restrict__ Wb, float* __restrict__ beta, int M) {
    __shared__ alignas(16) float sWb[3 * HD];
#if HAVE_TDM
    if (threadIdx.x < 32) {                 // wave 0 issues the TDM descriptor
        tdm_load_1d_f32(0u, Wb, 3 * HD);
        __builtin_amdgcn_s_wait_tensorcnt(0);
    }
#else
    for (int i = threadIdx.x; i < 3 * HD; i += 256) sWb[i] = Wb[i];
#endif
    __syncthreads();
    int m = blockIdx.x * blockDim.x + threadIdx.x;
    if (m >= M) return;
    const float* op = out + (size_t)m * HD;
    const float* xp = xr + (size_t)m * HD;
    float acc = 0.0f;
    for (int k = 0; k < HD; ++k) {
        float ov = op[k], xv = xp[k];
        acc += ov * sWb[k] + xv * sWb[HD + k] + (ov - xv) * sWb[2 * HD + k];
    }
    beta[m] = sigm_(acc);
}

__global__ void tf_combine_kernel(const float* __restrict__ out, const float* __restrict__ xr,
                                  const float* __restrict__ beta, float* __restrict__ hn,
                                  long long n) {
    long long i = (long long)blockIdx.x * blockDim.x + threadIdx.x;
    if (i >= n) return;
    int m = (int)(i >> 8);
    float b = beta[m];
    hn[i] = b * xr[i] + (1.0f - b) * out[i];
}

// -------------------- Set2Set --------------------
__global__ void lstm_gates_kernel(const float* __restrict__ qstar, const float* __restrict__ hx,
                                  const float* __restrict__ Wih, const float* __restrict__ Whh,
                                  const float* __restrict__ bih, const float* __restrict__ bhh,
                                  float* __restrict__ gates) {
    int idx = blockIdx.x * blockDim.x + threadIdx.x;   // NGRAPHS * 4H
    if (idx >= NGRAPHS * 4 * HD) return;
    int b = idx >> 10, j = idx & 1023;
    float acc = bih[j] + bhh[j];
    const float* qp = qstar + (size_t)b * (2 * HD);
    const float* wp = Wih + (size_t)j * (2 * HD);
    for (int k = 0; k < 2 * HD; ++k) acc += qp[k] * wp[k];
    const float* hp = hx + (size_t)b * HD;
    const float* up = Whh + (size_t)j * HD;
    for (int k = 0; k < HD; ++k) acc += hp[k] * up[k];
    gates[idx] = acc;
}

__global__ void lstm_update_kernel(const float* __restrict__ gates, float* __restrict__ hx,
                                   float* __restrict__ cx) {
    int idx = blockIdx.x * blockDim.x + threadIdx.x;   // NGRAPHS * H
    if (idx >= NGRAPHS * HD) return;
    int b = idx >> 8, j = idx & 255;
    const float* g = gates + (size_t)b * (4 * HD);
    float ig = sigm_(g[j]);
    float fg = sigm_(g[HD + j]);
    float gg = tanhf(g[2 * HD + j]);
    float og = sigm_(g[3 * HD + j]);
    float c = fg * cx[idx] + ig * gg;
    cx[idx] = c;
    hx[idx] = og * tanhf(c);
}

__global__ void s2s_score_kernel(const float* __restrict__ h, const float* __restrict__ hx,
                                 const int* __restrict__ batch, float* __restrict__ score) {
    int n = blockIdx.x * blockDim.x + threadIdx.x;
    if (n >= NNODES) return;
    int g = batch[n];
    const float4* hp = (const float4*)(h + (size_t)n * HD);
    const float4* xp = (const float4*)(hx + (size_t)g * HD);
    float acc = 0.0f;
    for (int k = 0; k < HD / 4; ++k) {
        float4 a = hp[k], bb = xp[k];
        acc += a.x * bb.x + a.y * bb.y + a.z * bb.z + a.w * bb.w;
    }
    score[n] = acc;
}

__global__ void s2s_segmax_kernel(const float* __restrict__ score, const int* __restrict__ batch,
                                  float* __restrict__ mG) {
    int n = blockIdx.x * blockDim.x + threadIdx.x;
    if (n >= NNODES) return;
    atomicMaxF(&mG[batch[n]], score[n]);
}

__global__ void s2s_exp_kernel(float* __restrict__ score, const int* __restrict__ batch,
                               const float* __restrict__ mG, float* __restrict__ denG) {
    int n = blockIdx.x * blockDim.x + threadIdx.x;
    if (n >= NNODES) return;
    float ex = __expf(score[n] - mG[batch[n]]);
    score[n] = ex;
    atomicAdd(&denG[batch[n]], ex);
}

__global__ void s2s_r_kernel(const float* __restrict__ h, const float* __restrict__ exS,
                             const float* __restrict__ denG, const int* __restrict__ batch,
                             float* __restrict__ r) {
    long long idx = (long long)blockIdx.x * blockDim.x + threadIdx.x;
    if (idx >= (long long)NNODES * (HD / 4)) return;
    int n = (int)(idx >> 6);
    int c4 = (int)(idx & 63) << 2;
    int g = batch[n];
    float a = exS[n] / (denG[g] + 1e-16f);
    float4 hv = *(const float4*)(h + (size_t)n * HD + c4);
    float* rp = r + (size_t)g * HD + c4;
    atomicAdd(rp + 0, a * hv.x);
    atomicAdd(rp + 1, a * hv.y);
    atomicAdd(rp + 2, a * hv.z);
    atomicAdd(rp + 3, a * hv.w);
}

__global__ void s2s_qstar_kernel(const float* __restrict__ hx, const float* __restrict__ r,
                                 float* __restrict__ qstar) {
    int idx = blockIdx.x * blockDim.x + threadIdx.x;   // NGRAPHS * 2H
    if (idx >= NGRAPHS * 2 * HD) return;
    int b = idx >> 9, j = idx & 511;
    qstar[idx] = (j < HD) ? hx[(size_t)b * HD + j] : r[(size_t)b * HD + (j - HD)];
}

// -------------------- readout tail --------------------
__global__ void small_linear_kernel(const float* __restrict__ A, const float* __restrict__ W,
                                    const float* __restrict__ b, const float* __restrict__ s,
                                    const float* __restrict__ o, float* __restrict__ C,
                                    int M, int K, int N, int mode) {
    int idx = blockIdx.x * blockDim.x + threadIdx.x;
    if (idx >= M * N) return;
    int m = idx / N, n = idx % N;
    float acc = b ? b[n] : 0.0f;
    const float* ap = A + (size_t)m * K;
    for (int k = 0; k < K; ++k) acc += ap[k] * W[(size_t)k * N + n];
    if (mode == EPI_BN_SILU) { acc = acc * s[n] + o[n]; acc = silu_(acc); }
    C[idx] = acc;
}

__global__ void concat2_kernel(const float* __restrict__ a, int na, const float* __restrict__ b,
                               int nb, float* __restrict__ c, int M) {
    int idx = blockIdx.x * blockDim.x + threadIdx.x;
    int tot = na + nb;
    if (idx >= M * tot) return;
    int m = idx / tot, j = idx % tot;
    c[idx] = (j < na) ? a[(size_t)m * na + j] : b[(size_t)m * nb + (j - na)];
}

__global__ void gf_kernel(const float* __restrict__ nA, const float* __restrict__ nB,
                          const float* __restrict__ ss, float* __restrict__ gf) {
    int b = blockIdx.x * blockDim.x + threadIdx.x;
    if (b >= NGRAPHS) return;
    gf[b * 2 + 0] = nA[b] / (ss[b] + 1e-10f);
    gf[b * 2 + 1] = nB[b] / (ss[b] + 1e-10f);
}

__global__ void final_out_kernel(const float* __restrict__ z2, const float* __restrict__ W3,
                                 const float* __restrict__ b3, float* __restrict__ out) {
    int b = blockIdx.x * blockDim.x + threadIdx.x;
    if (b >= NGRAPHS) return;
    float acc = b3[0];
    for (int k = 0; k < HD / 2; ++k) acc += z2[(size_t)b * (HD / 2) + k] * W3[k];
    out[b] = (acc > 20.0f) ? acc : log1pf(__expf(acc));   // softplus
}

// ---------------------------------------------------------------------------
// Host orchestration
// ---------------------------------------------------------------------------
extern "C" void kernel_launch(void* const* d_in, const int* in_sizes, int n_in,
                              void* d_out, int out_size, void* d_ws, size_t ws_size,
                              hipStream_t stream) {
    (void)in_sizes; (void)out_size; (void)ws_size;
    auto F = [&](int i) -> const float* { return (const float*)d_in[i]; };

    // top-level inputs (setup_inputs insertion order)
    const float* xIn  = F(0);
    const float* eaIn = F(1);
    const float* nAIn = F(2);
    const float* nBIn = F(3);
    const float* ssIn = F(4);
    // params pytree flattened with dict keys sorted (JAX convention), leaves 5..
    // edge_index / batch are the trailing integer inputs
    const int* ei    = (const int*)d_in[n_in - 2];
    const int* srcI  = ei;
    const int* dstI  = ei + NEDGES;
    const int* batch = (const int*)d_in[n_in - 1];

    // workspace bump allocator
    char* wp = (char*)d_ws;
    auto alloc = [&](size_t nfloats) -> float* {
        float* r = (float*)wp;
        wp += (nfloats * sizeof(float) + 255) & ~(size_t)255;
        return r;
    };
    float* e0    = alloc((size_t)NEDGES * HD);
    float* e1    = alloc((size_t)NEDGES * HD);
    float* eT    = alloc((size_t)NEDGES * HD);     // att.l1 temp, then eh
    float* wE    = alloc(NEDGES);
    float* alpha = alloc((size_t)NEDGES * NHEADS); // then ex, in-place
    float* h0    = alloc((size_t)NNODES * HD);
    float* h1    = alloc((size_t)NNODES * HD);
    float* hT    = alloc((size_t)NNODES * HD);
    float* hN    = alloc((size_t)NNODES * HD);
    float* agg   = alloc((size_t)NNODES * HD);     // also TF attention-out
    float* qB    = alloc((size_t)NNODES * HD);
    float* kB    = alloc((size_t)NNODES * HD);
    float* vB    = alloc((size_t)NNODES * HD);
    float* xrB   = alloc((size_t)NNODES * HD);
    float* mN    = alloc((size_t)NNODES * NHEADS);
    float* denN  = alloc((size_t)NNODES * NHEADS);
    float* betaB = alloc(NNODES);
    float* hx    = alloc(NGRAPHS * HD);
    float* cx    = alloc(NGRAPHS * HD);
    float* gates = alloc(NGRAPHS * 4 * HD);
    float* qs0   = alloc(NGRAPHS * 2 * HD);
    float* qs1   = alloc(NGRAPHS * 2 * HD);
    float* score = alloc(NNODES);
    float* mG    = alloc(NGRAPHS);
    float* denG  = alloc(NGRAPHS);
    float* rG    = alloc(NGRAPHS * HD);
    float* hr    = alloc(NGRAPHS * 4 * HD);
    float* hp    = alloc(NGRAPHS * 2 * HD);
    float* gfB   = alloc(NGRAPHS * 2);
    float* g1    = alloc(NGRAPHS * HD);
    float* goutB = alloc(NGRAPHS * HD);
    float* catB  = alloc(NGRAPHS * 3 * HD);
    float* z1    = alloc(NGRAPHS * HD);
    float* z2    = alloc(NGRAPHS * HD / 2);

    auto launch1d = [&](long long n) { return dim3((unsigned)((n + 255) / 256)); };
    auto gemm = [&](const float* A, const float* W, const float* bias, const float* s,
                    const float* o, const float* rs, float* C, int M, int N, int K, int mode) {
        dim3 g(M / GBM, N / GBN);
        if (rs) {
            if (mode == EPI_BN_SILU)
                wmma_gemm_t<true, EPI_BN_SILU><<<g, 256, 0, stream>>>(A, W, bias, s, o, rs, C, M, N, K);
            else
                wmma_gemm_t<true, EPI_NONE><<<g, 256, 0, stream>>>(A, W, bias, s, o, rs, C, M, N, K);
        } else {
            if (mode == EPI_BN_SILU)
                wmma_gemm_t<false, EPI_BN_SILU><<<g, 256, 0, stream>>>(A, W, bias, s, o, rs, C, M, N, K);
            else
                wmma_gemm_t<false, EPI_NONE><<<g, 256, 0, stream>>>(A, W, bias, s, o, rs, C, M, N, K);
        }
    };
    auto fill = [&](float* p, float v, long long n) {
        fill_kernel<<<launch1d(n), 256, 0, stream>>>(p, v, n);
    };

    // ---- encoders ----
    // node_enc leaves at 148: bn.o, bn.s, l.W, l.b
    enc_kernel<<<launch1d((long long)NNODES * HD), 256, 0, stream>>>(
        xIn, F(150), F(151), F(149), F(148), h0, NNODES, 4, HD);
    // edge_enc leaves at 5: bn.o, bn.s, l.W, l.b
    enc_kernel<<<launch1d((long long)NEDGES * HD), 256, 0, stream>>>(
        eaIn, F(7), F(8), F(6), F(5), e0, NEDGES, 3, HD);

    // ---- layers ----
    const int LB[6] = {25, 44, 66, 85, 107, 126};
    for (int i = 0; i < 6; ++i) {
        int B = LB[i];
        // att: bn.o,bn.s,l1.W,l1.b,l2.W,l2.b ; ec: bn.o,bn.s,l.W,l.b
        gemm(e0, F(B + 2), F(B + 3), F(B + 1), F(B + 0), nullptr, eT,
             NEDGES, HD, HD, EPI_BN_SILU);
        att_w_kernel<<<launch1d(NEDGES), 256, 0, stream>>>(eT, F(B + 4), F(B + 5), wE, NEDGES);
        gemm(e0, F(B + 8), F(B + 9), F(B + 7), F(B + 6), nullptr, e1,
             NEDGES, HD, HD, EPI_BN_SILU);
        std::swap(e0, e1);

        if ((i & 1) == 0) {
            // GINE: bn.o,bn.s,eps,l1.W,l1.b,l2.W,l2.b at B+10.. ; norm at B+17,B+18
            fill(agg, 0.0f, (long long)NNODES * HD);
            gine_scatter_kernel<<<launch1d((long long)NEDGES * 64), 256, 0, stream>>>(
                h0, e0, wE, srcI, dstI, agg);
            gine_combine_kernel<<<launch1d((long long)NNODES * HD), 256, 0, stream>>>(
                h0, agg, F(B + 12), hN, (long long)NNODES * HD);
            gemm(hN, F(B + 13), F(B + 14), F(B + 11), F(B + 10), nullptr, hT,
                 NNODES, HD, HD, EPI_BN_SILU);
            gemm(hT, F(B + 15), F(B + 16), nullptr, nullptr, nullptr, hN,
                 NNODES, HD, HD, EPI_NONE);
            resid_bn_kernel<<<launch1d((long long)NNODES * HD), 256, 0, stream>>>(
                h0, hN, F(B + 18), F(B + 17), (long long)NNODES * HD);
            // pool[i/2]: leaves 152 + 4*(i/2): bn.o, bn.s, l.W, l.b
            int P = 152 + 4 * (i / 2);
            gemm(h0, F(P + 2), F(P + 3), F(P + 1), F(P + 0), nullptr, h1,
                 NNODES, HD, HD, EPI_BN_SILU);
            std::swap(h0, h1);
        } else {
            // norm at B+10,B+11; tf: beta.W,e.W,k.W,k.b,q.W,q.b,skip.W,skip.b,v.W,v.b
            gemm(h0, F(B + 16), F(B + 17), nullptr, nullptr, nullptr, qB,
                 NNODES, HD, HD, EPI_NONE);
            gemm(h0, F(B + 14), F(B + 15), nullptr, nullptr, nullptr, kB,
                 NNODES, HD, HD, EPI_NONE);
            gemm(h0, F(B + 20), F(B + 21), nullptr, nullptr, nullptr, vB,
                 NNODES, HD, HD, EPI_NONE);
            gemm(h0, F(B + 18), F(B + 19), nullptr, nullptr, nullptr, xrB,
                 NNODES, HD, HD, EPI_NONE);
            // eh = (e * w) @ We  (rowscale fuses ew)
            gemm(e0, F(B + 13), nullptr, nullptr, nullptr, wE, eT,
                 NEDGES, HD, HD, EPI_NONE);
            tf_alpha_kernel<<<launch1d((long long)NEDGES * NHEADS), 256, 0, stream>>>(
                qB, kB, eT, srcI, dstI, alpha);
            fill(mN, -INFINITY, (long long)NNODES * NHEADS);
            fill(denN, 0.0f, (long long)NNODES * NHEADS);
            seg_max_kernel<<<launch1d((long long)NEDGES * NHEADS), 256, 0, stream>>>(
                alpha, dstI, mN);
            seg_exp_kernel<<<launch1d((long long)NEDGES * NHEADS), 256, 0, stream>>>(
                alpha, dstI, mN, denN);
            fill(agg, 0.0f, (long long)NNODES * HD);   // agg = attention out
            tf_out_kernel<<<launch1d((long long)NEDGES * 64), 256, 0, stream>>>(
                vB, eT, alpha, denN, srcI, dstI, agg);
            tf_beta_kernel<<<launch1d(NNODES), 256, 0, stream>>>(
                agg, xrB, F(B + 12), betaB, NNODES);
            tf_combine_kernel<<<launch1d((long long)NNODES * HD), 256, 0, stream>>>(
                agg, xrB, betaB, hN, (long long)NNODES * HD);
            resid_bn_kernel<<<launch1d((long long)NNODES * HD), 256, 0, stream>>>(
                h0, hN, F(B + 11), F(B + 10), (long long)NNODES * HD);
        }
    }

    // ---- Set2Set x2 (leaves 168+4j: Whh, Wih, bhh, bih) ----
    float* qsOut[2] = {qs0, qs1};
    for (int j = 0; j < 2; ++j) {
        const float* Whh = F(168 + 4 * j);
        const float* Wih = F(169 + 4 * j);
        const float* bhh = F(170 + 4 * j);
        const float* bih = F(171 + 4 * j);
        float* qstar = qsOut[j];
        fill(hx, 0.0f, NGRAPHS * HD);
        fill(cx, 0.0f, NGRAPHS * HD);
        fill(qstar, 0.0f, NGRAPHS * 2 * HD);
        for (int step = 0; step < 4; ++step) {
            lstm_gates_kernel<<<launch1d(NGRAPHS * 4 * HD), 256, 0, stream>>>(
                qstar, hx, Wih, Whh, bih, bhh, gates);
            lstm_update_kernel<<<launch1d(NGRAPHS * HD), 256, 0, stream>>>(gates, hx, cx);
            s2s_score_kernel<<<launch1d(NNODES), 256, 0, stream>>>(h0, hx, batch, score);
            fill(mG, -INFINITY, NGRAPHS);
            fill(denG, 0.0f, NGRAPHS);
            fill(rG, 0.0f, NGRAPHS * HD);
            s2s_segmax_kernel<<<launch1d(NNODES), 256, 0, stream>>>(score, batch, mG);
            s2s_exp_kernel<<<launch1d(NNODES), 256, 0, stream>>>(score, batch, mG, denG);
            s2s_r_kernel<<<launch1d((long long)NNODES * 64), 256, 0, stream>>>(
                h0, score, denG, batch, rG);
            s2s_qstar_kernel<<<launch1d(NGRAPHS * 2 * HD), 256, 0, stream>>>(hx, rG, qstar);
        }
    }
    concat2_kernel<<<launch1d(NGRAPHS * 4 * HD), 256, 0, stream>>>(
        qs0, 2 * HD, qs1, 2 * HD, hr, NGRAPHS);

    // ---- readout ----
    // proj leaves 164..167: bn.o, bn.s, l.W, l.b
    small_linear_kernel<<<launch1d(NGRAPHS * 2 * HD), 256, 0, stream>>>(
        hr, F(166), F(167), F(165), F(164), hp, NGRAPHS, 4 * HD, 2 * HD, EPI_BN_SILU);
    gf_kernel<<<1, 64, 0, stream>>>(nAIn, nBIn, ssIn, gfB);
    // gmlp leaves 19..24: bn.o, bn.s, l1.W, l1.b, l2.W, l2.b
    small_linear_kernel<<<launch1d(NGRAPHS * HD), 256, 0, stream>>>(
        gfB, F(21), F(22), F(20), F(19), g1, NGRAPHS, 2, HD, EPI_BN_SILU);
    small_linear_kernel<<<launch1d(NGRAPHS * HD), 256, 0, stream>>>(
        g1, F(23), F(24), nullptr, nullptr, goutB, NGRAPHS, HD, HD, EPI_NONE);
    concat2_kernel<<<launch1d(NGRAPHS * 3 * HD), 256, 0, stream>>>(
        hp, 2 * HD, goutB, HD, catB, NGRAPHS);
    // final leaves: bn1.o=9 bn1.s=10 bn2.o=11 bn2.s=12 l1.W=13 l1.b=14 l2.W=15 l2.b=16 l3.W=17 l3.b=18
    small_linear_kernel<<<launch1d(NGRAPHS * HD), 256, 0, stream>>>(
        catB, F(13), F(14), F(10), F(9), z1, NGRAPHS, 3 * HD, HD, EPI_BN_SILU);
    small_linear_kernel<<<launch1d(NGRAPHS * HD / 2), 256, 0, stream>>>(
        z1, F(15), F(16), F(12), F(11), z2, NGRAPHS, HD, HD / 2, EPI_BN_SILU);
    final_out_kernel<<<1, 64, 0, stream>>>(z2, F(17), F(18), (float*)d_out);
}
